// WACBlock_57758720197091
// MI455X (gfx1250) — matrix operand order repaired and checked by reading, not compiled, not run on hardware
//
#include <hip/hip_runtime.h>
#include <hip/hip_bf16.h>

typedef __bf16 bf16_t;
typedef __attribute__((ext_vector_type(16))) __bf16 v16bf;
typedef __attribute__((ext_vector_type(8)))  float  v8f;
typedef __attribute__((ext_vector_type(4)))  int    v4i;

#ifndef __has_builtin
#define __has_builtin(x) 0
#endif
#if __has_builtin(__builtin_amdgcn_global_load_async_to_lds_b128) && \
    __has_builtin(__builtin_amdgcn_s_wait_asynccnt)
#define USE_ASYNC_LDS 1
#else
#define USE_ASYNC_LDS 0
#endif

#if USE_ASYNC_LDS
typedef __attribute__((address_space(1))) void* as1_void;
typedef __attribute__((address_space(3))) void* as3_void;
typedef __attribute__((address_space(1))) v4i*  as1_v4i;
typedef __attribute__((address_space(3))) v4i*  as3_v4i;
// 16B global -> LDS direct copy, tracked by ASYNCcnt (no VGPR round trip)
__device__ __forceinline__ void async_copy16(const bf16_t* g, bf16_t* l) {
  __builtin_amdgcn_global_load_async_to_lds_b128(
      (as1_v4i)(as1_void)(void*)const_cast<bf16_t*>(g),
      (as3_v4i)(as3_void)(void*)l, /*offset=*/0, /*cpol=*/0);
}
#endif

// ---------------------------------------------------------------------------
// WMMA helpers (CDNA5 wave32, V_WMMA_F32_16X16X32_BF16)
// ---------------------------------------------------------------------------
__device__ __forceinline__ v8f wmma_bf16(v16bf a, v16bf b, v8f c) {
  return __builtin_amdgcn_wmma_f32_16x16x32_bf16(
      /*neg_a=*/false, a, /*neg_b=*/false, b,
      /*c_mod=*/(short)0, c, /*reuse_a=*/false, /*reuse_b=*/false);
}

// A 16x32 (MxK), row-major source: element (m,k) = p[m*ld + k]
// contiguous k order used here (matches B loaders): lanes 0-15 -> M=lane, K 0..15; lanes 16-31 -> K 16..31
__device__ __forceinline__ v16bf load_a_rm(const bf16_t* p, int ld) {
  int lane = threadIdx.x & 31;
  int m = lane & 15;
  int kb = (lane < 16) ? 0 : 16;
  const bf16_t* r = p + m * ld;
  v16bf a;
#pragma unroll
  for (int i = 0; i < 8; ++i) a[i] = r[kb + i];
#pragma unroll
  for (int i = 0; i < 8; ++i) a[8 + i] = r[kb + 8 + i];
  return a;
}

// A 16x32 with only K=0..15 valid (K 16..31 zero padded), row-major source
__device__ __forceinline__ v16bf load_a_k16(const bf16_t* p, int ld) {
  int lane = threadIdx.x & 31;
  int m = lane & 15;
  bool lo = lane < 16;
  const bf16_t* r = p + m * ld;
  v16bf a;
#pragma unroll
  for (int i = 0; i < 8; ++i) a[i] = lo ? r[i] : (bf16_t)0.0f;
#pragma unroll
  for (int i = 0; i < 8; ++i) a[8 + i] = lo ? r[8 + i] : (bf16_t)0.0f;
  return a;
}

// B 32x16 (KxN), K-major source: element (k,n) = p[k*ld + n]  (strided; cold paths only)
__device__ __forceinline__ v16bf load_b_rm(const bf16_t* p, int ld) {
  int lane = threadIdx.x & 31;
  int n = lane & 15;
  int kb = (lane < 16) ? 0 : 16;
  v16bf b;
#pragma unroll
  for (int i = 0; i < 16; ++i) b[i] = p[(kb + i) * ld + n];
  return b;
}

// B 32x16 from transposed storage (stored [n][k]): element (k,n) = p[n*ld + k]
// contiguous 32B per lane -> 2x b128 loads
__device__ __forceinline__ v16bf load_b_tr(const bf16_t* p, int ld) {
  int lane = threadIdx.x & 31;
  int n = lane & 15;
  int kb = (lane < 16) ? 0 : 16;
  const bf16_t* r = p + n * ld;
  v16bf b;
#pragma unroll
  for (int i = 0; i < 16; ++i) b[i] = r[kb + i];
  return b;
}

// B 32x16 transposed storage, only K=0..15 valid (rest zero)
__device__ __forceinline__ v16bf load_b_tr_k16(const bf16_t* p, int ld) {
  int lane = threadIdx.x & 31;
  int n = lane & 15;
  v16bf b;
  if (lane < 16) {
    const bf16_t* r = p + n * ld;
#pragma unroll
    for (int i = 0; i < 16; ++i) b[i] = r[i];
  } else {
#pragma unroll
    for (int i = 0; i < 16; ++i) b[i] = (bf16_t)0.0f;
  }
  return b;
}

// ---------------------------------------------------------------------------
// Kernel 0: relative-position bias MLP + gather  -> bias2d[64*64] f32
// ---------------------------------------------------------------------------
__global__ void k_bias(const float* __restrict__ b1w, const float* __restrict__ b1b,
                       const float* __restrict__ b2w, const float* __restrict__ b2b,
                       const float* __restrict__ bd, const int* __restrict__ bidx,
                       float* __restrict__ bias2d, int U) {
  __shared__ float val[256];
  int t = threadIdx.x;
  if (t < U) {
    float d0 = bd[t * 2], d1 = bd[t * 2 + 1];
    float acc = b2b[0];
#pragma unroll
    for (int j = 0; j < 16; ++j) {
      float h = d0 * b1w[j * 2] + d1 * b1w[j * 2 + 1] + b1b[j];
      h = 0.5f * h * (1.0f + erff(h * 0.70710678118654752f));  // exact GELU
      acc += b2w[j] * h;
    }
    val[t] = acc;
  }
  __syncthreads();
  for (int i = t; i < 4096; i += 256) bias2d[i] = val[bidx[i]];
}

// f32 -> bf16 weight conversion
__global__ void k_cvt(const float* __restrict__ in, bf16_t* __restrict__ out, int n) {
  int i = blockIdx.x * 256 + threadIdx.x;
  if (i < n) out[i] = (bf16_t)in[i];
}

// cw2 [co][ci][ky][kx] f32 -> [tap][co][ci] bf16  (so B loads are contiguous)
__global__ void k_cvt_cw2(const float* __restrict__ in, bf16_t* __restrict__ out) {
  int i = blockIdx.x * 256 + threadIdx.x;
  if (i < 192 * 192 * 9) {
    int co = i / (192 * 9);
    int r = i % (192 * 9);
    int ci = r / 9;
    int tap = r % 9;
    out[(tap * 192 + co) * 192 + ci] = (bf16_t)in[i];
  }
}

// ---------------------------------------------------------------------------
// Kernel A: per-window LayerNorm + QKV + MHA(+bias) + proj + residual
// one workgroup (8 waves) per 8x8 window; writes x+attn into d_out (f32 BCHW)
// ---------------------------------------------------------------------------
__global__ void __launch_bounds__(256) k_attn(
    const float* __restrict__ x, const float* __restrict__ ln_w,
    const float* __restrict__ qkv_b, const float* __restrict__ proj_b,
    const bf16_t* __restrict__ qkvw, const bf16_t* __restrict__ projw,
    const float* __restrict__ bias2d, float* __restrict__ xattn) {
  __shared__ __align__(16) unsigned char smem[60160];
  bf16_t* xn   = (bf16_t*)smem;             // [64][200] bf16 (25600 B)
  bf16_t* qkv  = (bf16_t*)(smem + 25600);   // [64][200] bf16 (25600 B)
  bf16_t* ob   = (bf16_t*)(smem + 51200);   // [64][68]  bf16 ( 8704 B)
  float*  sc   = (float*)smem;              // [64][66]  f32  (reuses xn region)
  bf16_t* attn = (bf16_t*)(smem + 16896);   // [64][66]  bf16 (reuses xn region)

  int tid = threadIdx.x;
  int lane = tid & 31;
  int wave = tid >> 5;
  int nloc = lane & 15;
  int mbase = (lane >> 4) << 3;  // C/D layout: lanes>=16 hold M = r+8

  int b = blockIdx.y;
  int wy = blockIdx.x / 24, wx = blockIdx.x % 24;
  int y0 = wy * 8, x0 = wx * 8;

  // stage window into LDS as bf16, token-major [pix][c]
  for (int i = tid; i < 64 * 192; i += 256) {
    int c = i >> 6, pix = i & 63;
    int row = pix >> 3, col = pix & 7;
    float v = x[((b * 192 + c) * 192 + y0 + row) * 192 + x0 + col];
    xn[pix * 200 + c] = (bf16_t)v;
  }
  __syncthreads();

  // pre-norm LayerNorm (bias-free), one token per thread
  if (tid < 64) {
    float s = 0.f, s2 = 0.f;
    for (int c = 0; c < 192; ++c) { float v = (float)xn[tid * 200 + c]; s += v; s2 += v * v; }
    float mu = s * (1.f / 192.f);
    float var = s2 * (1.f / 192.f) - mu * mu;
    float inv = rsqrtf(var + 1e-5f);
    for (int c = 0; c < 192; ++c) {
      float v = (float)xn[tid * 200 + c];
      xn[tid * 200 + c] = (bf16_t)((v - mu) * inv * ln_w[c]);
    }
  }
  __syncthreads();

  int mt4 = wave & 3;
  int ntb = (wave >> 2) * 6;

  // QKV GEMM: wave owns M-tile mt4 and 6 N-tiles; A loaded once per k-step,
  // 3 independent WMMAs share it. q pre-scaled by 1/sqrt(16).
  for (int g = 0; g < 2; ++g) {
    int n0 = ntb + g * 3;
    v8f accs[3] = {{}, {}, {}};
#pragma unroll
    for (int kt = 0; kt < 6; ++kt) {
      v16bf a = load_a_rm(&xn[(mt4 * 16) * 200 + kt * 32], 200);
#pragma unroll
      for (int j = 0; j < 3; ++j)
        accs[j] = wmma_bf16(a, load_b_tr(&qkvw[((n0 + j) * 16) * 192 + kt * 32], 192), accs[j]);
    }
#pragma unroll
    for (int j = 0; j < 3; ++j) {
      int jg = (n0 + j) * 16 + nloc;
      float bv = qkv_b[jg];
      float scale = (jg < 64) ? 0.25f : 1.0f;
#pragma unroll
      for (int r = 0; r < 8; ++r) {
        int m = mt4 * 16 + mbase + r;
        qkv[m * 200 + jg] = (bf16_t)((accs[j][r] + bv) * scale);
      }
    }
  }
  __syncthreads();  // xn region now free -> scores/attn live there

  for (int h = 0; h < 4; ++h) {
    // scores = q @ k^T + bias (K=16 zero padded). Both of a wave's tiles share
    // the same N-tile, so k^T is loaded once per wave.
    {
      int nt = wave >> 1;
      int mt0 = (wave & 1) * 2;
      v16bf bb = load_b_tr_k16(&qkv[(nt * 16) * 200 + 64 + h * 16], 200);
#pragma unroll
      for (int j = 0; j < 2; ++j) {
        int mt = mt0 + j;
        v16bf a = load_a_k16(&qkv[(mt * 16) * 200 + h * 16], 200);
        v8f acc = {};
        acc = wmma_bf16(a, bb, acc);
#pragma unroll
        for (int r = 0; r < 8; ++r) {
          int m = mt * 16 + mbase + r;
          int n = nt * 16 + nloc;
          sc[m * 66 + n] = acc[r] + bias2d[m * 64 + n];
        }
      }
    }
    __syncthreads();

    // row softmax -> bf16 attn
    if (tid < 64) {
      float mx = -3.4e38f;
      for (int j = 0; j < 64; ++j) mx = fmaxf(mx, sc[tid * 66 + j]);
      float s = 0.f;
      for (int j = 0; j < 64; ++j) { float e = expf(sc[tid * 66 + j] - mx); sc[tid * 66 + j] = e; s += e; }
      float inv = 1.f / s;
      for (int j = 0; j < 64; ++j) attn[tid * 66 + j] = (bf16_t)(sc[tid * 66 + j] * inv);
    }
    __syncthreads();

    // o = attn @ v  (M=64, N=16, K=64)
    if (wave < 4) {
      int mt = wave;
      v8f acc = {};
#pragma unroll
      for (int kt = 0; kt < 2; ++kt) {
        v16bf a  = load_a_rm(&attn[(mt * 16) * 66 + kt * 32], 66);
        v16bf bb = load_b_rm(&qkv[(kt * 32) * 200 + 128 + h * 16], 200);
        acc = wmma_bf16(a, bb, acc);
      }
#pragma unroll
      for (int r = 0; r < 8; ++r) {
        int m = mt * 16 + mbase + r;
        ob[m * 68 + h * 16 + nloc] = (bf16_t)acc[r];
      }
    }
    __syncthreads();
  }

  // proj [64,64] x proj_w^T [64,192] + residual -> global f32
  for (int g = 0; g < 2; ++g) {
    int n0 = ntb + g * 3;
    v8f accs[3] = {{}, {}, {}};
#pragma unroll
    for (int kt = 0; kt < 2; ++kt) {
      v16bf a = load_a_rm(&ob[(mt4 * 16) * 68 + kt * 32], 68);
#pragma unroll
      for (int j = 0; j < 3; ++j)
        accs[j] = wmma_bf16(a, load_b_tr(&projw[((n0 + j) * 16) * 64 + kt * 32], 64), accs[j]);
    }
#pragma unroll
    for (int j = 0; j < 3; ++j) {
      int cg = (n0 + j) * 16 + nloc;
      float pb = proj_b[cg];
#pragma unroll
      for (int r = 0; r < 8; ++r) {
        int m = mt4 * 16 + mbase + r;
        int row = m >> 3, col = m & 7;
        int gi = ((b * 192 + cg) * 192 + y0 + row) * 192 + x0 + col;
        xattn[gi] = accs[j][r] + pb + x[gi];
      }
    }
  }
}

// ---------------------------------------------------------------------------
// Kernel B: 1x1 conv C->2C + GLU. 64-pixel row tiles. Output bf16 in
// channel-fastest layout [b][h][w][c] via LDS transpose (coalesced stores).
// ---------------------------------------------------------------------------
__global__ void __launch_bounds__(256) k_glu(
    const float* __restrict__ xattn, const bf16_t* __restrict__ w1,
    const float* __restrict__ cb1, bf16_t* __restrict__ y) {
  __shared__ __align__(16) bf16_t xin[64 * 200];   // pixel-major [p][c], 25600 B
  __shared__ __align__(16) bf16_t yout[64 * 200];  // GLU result  [p][c], 25600 B
  int tid = threadIdx.x;
  int lane = tid & 31, wave = tid >> 5;
  int nloc = lane & 15, mbase = (lane >> 4) << 3;
  int b = blockIdx.y;
  int h = blockIdx.x / 3;
  int x0 = (blockIdx.x % 3) * 64;

  for (int i = tid; i < 192 * 64; i += 256) {
    int c = i >> 6, p = i & 63;  // p fastest: coalesced global reads
    xin[p * 200 + c] = (bf16_t)xattn[((b * 192 + c) * 192 + h) * 192 + x0 + p];
  }
  __syncthreads();

  int mt = wave & 3;
  int npb = (wave >> 2) * 6;
  for (int g = 0; g < 2; ++g) {
    int np0 = npb + g * 3;
    v8f accA[3] = {{}, {}, {}}, accB[3] = {{}, {}, {}};
#pragma unroll
    for (int kt = 0; kt < 6; ++kt) {
      v16bf a = load_a_rm(&xin[(mt * 16) * 200 + kt * 32], 200);
#pragma unroll
      for (int j = 0; j < 3; ++j) {
        accA[j] = wmma_bf16(a, load_b_tr(&w1[((np0 + j) * 16) * 192 + kt * 32], 192), accA[j]);
        accB[j] = wmma_bf16(a, load_b_tr(&w1[(192 + (np0 + j) * 16) * 192 + kt * 32], 192), accB[j]);
      }
    }
#pragma unroll
    for (int j = 0; j < 3; ++j) {
      int c = (np0 + j) * 16 + nloc;
      float ba = cb1[c], bg = cb1[192 + c];
#pragma unroll
      for (int r = 0; r < 8; ++r) {
        int m = mt * 16 + mbase + r;
        float av = accA[j][r] + ba;
        float bv = accB[j][r] + bg;
        yout[m * 200 + c] = (bf16_t)(av / (1.f + expf(-bv)));  // a * sigmoid(b)
      }
    }
  }
  __syncthreads();

  // coalesced store, channel-fastest: y[((b*H + h)*W + w)*C + c]
  for (int i = tid; i < 64 * 192; i += 256) {
    int p = i / 192, c = i % 192;
    y[((b * 192 + h) * 192 + x0 + p) * 192 + c] = yout[p * 200 + c];
  }
}

// ---------------------------------------------------------------------------
// Kernel C: 3x3 conv (replicate pad) + LeakyReLU(0.1) + residual (in-place d_out)
// 2 output rows x 32 cols per WG (4-row halo staged once -> y read only 2x).
// Halo staged via GLOBAL_LOAD_ASYNC_TO_LDS_B128 when the toolchain exposes it.
// ---------------------------------------------------------------------------
__global__ void __launch_bounds__(256) k_conv3(
    const bf16_t* __restrict__ y, const bf16_t* __restrict__ w2t,
    const float* __restrict__ cb2, float* __restrict__ out) {
  __shared__ __align__(16) bf16_t yl[4 * 34 * 200];  // [row][p][c], 54400 B
  int tid = threadIdx.x;
  int lane = tid & 31, wave = tid >> 5;
  int nloc = lane & 15, mbase = (lane >> 4) << 3;
  int b = blockIdx.y;
  int h0 = (blockIdx.x / 6) * 2;         // two output rows h0, h0+1
  int x0 = (blockIdx.x % 6) * 32;

  // stage 4-row halo in 16B chunks (8 bf16 channels per chunk); both sides are
  // contiguous in c, so this is a pure LDS-direct copy.
  for (int ch = tid; ch < 4 * 34 * 24; ch += 256) {
    int c = (ch % 24) * 8;
    int pp = ch / 24;
    int row = pp / 34, p = pp % 34;
    int gr = min(max(h0 - 1 + row, 0), 191);   // replicate pad (rows)
    int gx = min(max(x0 - 1 + p, 0), 191);     // replicate pad (cols)
    const bf16_t* gaddr = &y[((b * 192 + gr) * 192 + gx) * 192 + c];
    bf16_t* laddr = &yl[(row * 34 + p) * 200 + c];
#if USE_ASYNC_LDS
    async_copy16(gaddr, laddr);
#else
    *(uint4*)laddr = *(const uint4*)gaddr;
#endif
  }
#if USE_ASYNC_LDS
  __builtin_amdgcn_s_wait_asynccnt(0);
#endif
  __syncthreads();

  // M = 64 pixels = 2 rows x 32 cols; wave owns 1 M-tile and 6 N-tiles
  int mt = wave & 3;
  int r0 = mt >> 1;                 // output row within tile pair
  int p0 = (mt & 1) * 16;           // column base within 32-wide strip
  int ntb = (wave >> 2) * 6;
  for (int g = 0; g < 2; ++g) {
    int nt0 = ntb + g * 3;
    v8f accs[3] = {{}, {}, {}};
    for (int tap = 0; tap < 9; ++tap) {
      int ky = tap / 3, kx = tap % 3;
      const bf16_t* arow = &yl[((r0 + ky) * 34 + p0 + kx) * 200];
      const bf16_t* wtap = &w2t[(tap * 192) * 192];
      if (tap < 8)  // pull next tap's weight rows toward the caches
        __builtin_prefetch(wtap + 192 * 192 + (nt0 * 16) * 192, 0, 1);
#pragma unroll
      for (int kt = 0; kt < 6; ++kt) {
        v16bf a = load_a_rm(arow + kt * 32, 200);
#pragma unroll
        for (int j = 0; j < 3; ++j)
          accs[j] = wmma_bf16(a, load_b_tr(wtap + ((nt0 + j) * 16) * 192 + kt * 32, 192), accs[j]);
      }
    }
#pragma unroll
    for (int j = 0; j < 3; ++j) {
      int c = (nt0 + j) * 16 + nloc;
      float bv = cb2[c];
#pragma unroll
      for (int r = 0; r < 8; ++r) {
        int m = mbase + r;
        float tv = accs[j][r] + bv;
        tv = (tv >= 0.f) ? tv : 0.1f * tv;
        int gi = ((b * 192 + c) * 192 + h0 + r0) * 192 + x0 + p0 + m;
        out[gi] = out[gi] + tv;   // residual: out currently holds x+attn
      }
    }
  }
}

// ---------------------------------------------------------------------------
extern "C" void kernel_launch(void* const* d_in, const int* in_sizes, int n_in,
                              void* d_out, int out_size, void* d_ws, size_t ws_size,
                              hipStream_t stream) {
  const float* x      = (const float*)d_in[0];
  const float* ln_w   = (const float*)d_in[1];
  const float* qkv_w  = (const float*)d_in[2];
  const float* qkv_b  = (const float*)d_in[3];
  const float* proj_w = (const float*)d_in[4];
  const float* proj_b = (const float*)d_in[5];
  const float* b1_w   = (const float*)d_in[6];
  const float* b1_b   = (const float*)d_in[7];
  const float* b2_w   = (const float*)d_in[8];
  const float* b2_b   = (const float*)d_in[9];
  const float* cw1    = (const float*)d_in[10];
  const float* cb1    = (const float*)d_in[11];
  const float* cw2    = (const float*)d_in[12];
  const float* cb2    = (const float*)d_in[13];
  const int*   bidx   = (const int*)d_in[14];
  const float* bdelta = (const float*)d_in[15];
  int U = in_sizes[15] / 2;

  char* ws = (char*)d_ws;
  float*  bias2d = (float*)(ws + 0);            //  16384 B
  bf16_t* qkvwb  = (bf16_t*)(ws + 16384);       //  73728 B (192*192)
  bf16_t* projwb = (bf16_t*)(ws + 90112);       //  24576 B (192*64)
  bf16_t* cw1b   = (bf16_t*)(ws + 114688);      // 147456 B (384*192)
  bf16_t* w2t    = (bf16_t*)(ws + 262144);      // 663552 B (9*192*192, [tap][co][ci])
  bf16_t* yws    = (bf16_t*)(ws + 925696);      // 113246208 B (8*192*192*192, [b][h][w][c])
  float*  xattn  = (float*)d_out;               // d_out doubles as f32 intermediate

  // prep: bias table + weight precision conversion
  k_bias<<<1, 256, 0, stream>>>(b1_w, b1_b, b2_w, b2_b, bdelta, bidx, bias2d, U);
  k_cvt<<<(36864 + 255) / 256, 256, 0, stream>>>(qkv_w, qkvwb, 36864);
  k_cvt<<<(12288 + 255) / 256, 256, 0, stream>>>(proj_w, projwb, 12288);
  k_cvt<<<(73728 + 255) / 256, 256, 0, stream>>>(cw1, cw1b, 73728);
  k_cvt_cw2<<<(331776 + 255) / 256, 256, 0, stream>>>(cw2, w2t);

  // windowed attention block -> d_out = x + attn(x)
  k_attn<<<dim3(576, 8), 256, 0, stream>>>(x, ln_w, qkv_b, proj_b,
                                           qkvwb, projwb, bias2d, xattn);
  // 1x1 conv + GLU -> yws (bf16, channel-fastest)
  k_glu<<<dim3(576, 8), 256, 0, stream>>>(xattn, cw1b, cb1, yws);
  // 3x3 conv + LeakyReLU + residual (in place on d_out); 2 rows per WG
  k_conv3<<<dim3(576, 8), 256, 0, stream>>>(yws, w2t, cb2, xattn);
}
